// MultiHeadSelfAttention_13451837571140
// MI455X (gfx1250) — matrix operand order repaired and checked by reading, not compiled, not run on hardware
//
#include <hip/hip_runtime.h>
#include <hip/hip_bf16.h>
#include <stdint.h>

// ---------------------------------------------------------------------------
// MHA: B=2, N=2048, D=1024, H=16, HD=64.  bf16 WMMA (16x16x32) everywhere,
// f32 accumulate, flash-attention so the 512MB score tensor never exists.
// Round 3: even-trip-count pipelined loops + #pragma unroll 2 so the
// cur/nxt fragment rotation copy-propagates away (ping-pong register sets,
// no v_mov rotation, waits stay partial across iterations).
// ---------------------------------------------------------------------------

#define BB 2
#define NN 2048
#define DD 1024
#define HH 16
#define HD 64
#define SCALE 0.125f  // 1/sqrt(64)

typedef __attribute__((ext_vector_type(16))) __bf16 v16bf;
typedef __attribute__((ext_vector_type(8)))  float  v8f;

union FragU {
    uint4 q[2];
    v16bf v;
};

__device__ __forceinline__ __bf16 f2bf(float f) {
    union { float f; uint32_t u; } v;
    v.f = f;
    uint32_t u = v.u;
    uint32_t r = u + 0x7FFFu + ((u >> 16) & 1u);   // round-to-nearest-even
    unsigned short h = (unsigned short)(r >> 16);
    return __builtin_bit_cast(__bf16, h);
}

__device__ __forceinline__ v8f wmma_bf16(v16bf a, v16bf b, v8f c) {
    return __builtin_amdgcn_wmma_f32_16x16x32_bf16(false, a, false, b, (short)0, c,
                                                   false, false);
}

// A fragment (16x32 bf16, M x K): lane (m = lane&15, hi = lane>>4) holds
// K = k0 + 16*hi + [0..15].
__device__ __forceinline__ v16bf load_afrag(const __bf16* base, int lda, int m0, int k0,
                                            int lane) {
    int m  = lane & 15;
    int hi = lane >> 4;
    const __bf16* p = base + (size_t)(m0 + m) * lda + k0 + 16 * hi;
    FragU f;
    f.q[0] = *(const uint4*)(p);
    f.q[1] = *(const uint4*)(p + 8);
    return f.v;
}

// B fragment (32x16 bf16, K x N) from a K-contiguous ("B^T") layout:
// lane (n = lane&15, hi = lane>>4) holds K = k0 + 16*hi + [0..15].
__device__ __forceinline__ v16bf load_bfrag(const __bf16* baseT, int ldb, int n0, int k0,
                                            int lane) {
    int n  = lane & 15;
    int hi = lane >> 4;
    const __bf16* p = baseT + (size_t)(n0 + n) * ldb + k0 + 16 * hi;
    FragU f;
    f.q[0] = *(const uint4*)(p);
    f.q[1] = *(const uint4*)(p + 8);
    return f.v;
}

// ---------------------------------------------------------------------------
// Kernel 1: convert x -> bf16; transpose+convert w_qkv -> [3D,D] bf16,
// w_out -> [D,D]^T bf16 (so all GEMM B-operands are K-contiguous rows).
// ---------------------------------------------------------------------------
#define NX   (size_t)(BB * NN * DD)          // 4194304
#define NW1  (size_t)(DD * 3 * DD)           // 3145728
#define NW2  (size_t)(DD * DD)               // 1048576

__global__ __launch_bounds__(256) void convert_kernel(const float* __restrict__ x,
                                                      const float* __restrict__ wqkv,
                                                      const float* __restrict__ wout,
                                                      __bf16* __restrict__ xbf,
                                                      __bf16* __restrict__ wqkvT,
                                                      __bf16* __restrict__ woutT) {
    size_t idx = (size_t)blockIdx.x * blockDim.x + threadIdx.x;
    if (idx < NX) {
        xbf[idx] = f2bf(x[idx]);
    } else if (idx < NX + NW1) {
        size_t t = idx - NX;
        int n = (int)(t >> 10);
        int k = (int)(t & 1023);
        wqkvT[t] = f2bf(wqkv[(size_t)k * (3 * DD) + n]);
    } else if (idx < NX + NW1 + NW2) {
        size_t t = idx - NX - NW1;
        int n = (int)(t >> 10);
        int k = (int)(t & 1023);
        woutT[t] = f2bf(wout[(size_t)k * DD + n]);
    }
}

// ---------------------------------------------------------------------------
// Pipelined 16x64 GEMM core: acc[4] += A[row0:16, :K] * Bt[n0:64, :K]^T
// 32 uniform steps; step i issues loads for step i+1 (clamped on the last
// step) before consuming step i's fragments.  Unroll-2 -> ping-pong regs.
// ---------------------------------------------------------------------------
__device__ __forceinline__ void gemm_16x64_pipelined(const __bf16* __restrict__ A,
                                                     const __bf16* __restrict__ Bt,
                                                     int row0, int n0, int lane,
                                                     v8f acc[4]) {
    v16bf a_cur = load_afrag(A, DD, row0, 0, lane);
    v16bf b_cur[4];
#pragma unroll
    for (int t = 0; t < 4; ++t) b_cur[t] = load_bfrag(Bt, DD, n0 + 16 * t, 0, lane);

#pragma unroll 2
    for (int k0 = 0; k0 < DD; k0 += 32) {
        int kn = (k0 + 32 < DD) ? (k0 + 32) : k0;   // last step: harmless reload
        v16bf a_nxt = load_afrag(A, DD, row0, kn, lane);
        v16bf b_nxt[4];
#pragma unroll
        for (int t = 0; t < 4; ++t) b_nxt[t] = load_bfrag(Bt, DD, n0 + 16 * t, kn, lane);

#pragma unroll
        for (int t = 0; t < 4; ++t) acc[t] = wmma_bf16(a_cur, b_cur[t], acc[t]);

        a_cur = a_nxt;
#pragma unroll
        for (int t = 0; t < 4; ++t) b_cur[t] = b_nxt[t];
    }
}

// ---------------------------------------------------------------------------
// Kernel 2: QKV GEMM.  C[4096,3072] = Xbf[4096,1024] @ Wqkv + b.
// Epilogue scatters Q,K -> [B,H,N,HD], V -> [B,H,HD,N] (transposed).
// ---------------------------------------------------------------------------
__global__ __launch_bounds__(256) void qkv_gemm_kernel(const __bf16* __restrict__ X,
                                                       const __bf16* __restrict__ Wt,
                                                       const float* __restrict__ bias,
                                                       __bf16* __restrict__ Qo,
                                                       __bf16* __restrict__ Ko,
                                                       __bf16* __restrict__ Vto) {
    int wid  = blockIdx.x * 8 + (threadIdx.x >> 5);
    int lane = threadIdx.x & 31;
    int mt = wid / 48;              // 256 row-tiles of 16
    int nt = wid % 48;              // 48 col-tiles of 64
    int row0 = mt * 16;
    int n0   = nt * 64;

    v8f acc[4] = {v8f{}, v8f{}, v8f{}, v8f{}};
    gemm_16x64_pipelined(X, Wt, row0, n0, lane, acc);

    int col   = lane & 15;
    int hflag = lane >> 4;
#pragma unroll
    for (int t = 0; t < 4; ++t) {
        int   c  = n0 + 16 * t + col;
        float bv = bias[c];
#pragma unroll
        for (int i = 0; i < 8; ++i) {
            int    r  = row0 + i + 8 * hflag;       // global token row (0..4095)
            float  v  = acc[t][i] + bv;
            __bf16 bb = f2bf(v);
            int bidx = r >> 11;                     // / N
            int n    = r & (NN - 1);
            if (c < DD) {                           // Q
                int hh = c >> 6, hd = c & 63;
                Qo[(((size_t)bidx * HH + hh) * NN + n) * HD + hd] = bb;
            } else if (c < 2 * DD) {                // K
                int c2 = c - DD;
                int hh = c2 >> 6, hd = c2 & 63;
                Ko[(((size_t)bidx * HH + hh) * NN + n) * HD + hd] = bb;
            } else {                                // V (transposed)
                int c2 = c - 2 * DD;
                int hh = c2 >> 6, hd = c2 & 63;
                Vto[(((size_t)bidx * HH + hh) * HD + hd) * NN + n] = bb;
            }
        }
    }
}

// ---------------------------------------------------------------------------
// Kernel 3: flash attention.  One wave per (b,h, 16-row q-tile).
// Pipelined + unroll-2: V-frags for step j and K-frags for step j+1 are in
// flight while the softmax VALU/TRANS work executes.
// ---------------------------------------------------------------------------
__global__ __launch_bounds__(32) void attn_kernel(const __bf16* __restrict__ Q,
                                                  const __bf16* __restrict__ Kt,
                                                  const __bf16* __restrict__ Vt,
                                                  __bf16* __restrict__ ctx) {
    __shared__ __align__(32) __bf16 Pl[16 * 32];

    int lane = threadIdx.x & 31;
    int w    = blockIdx.x;                 // 0..4095
    int qt   = w & 127;                    // N/16 tiles
    int bh   = w >> 7;                     // b*H + h
    int b    = bh >> 4;
    int h    = bh & 15;
    int row0 = qt * 16;

    const __bf16* Qb = Q  + (size_t)bh * NN * HD;
    const __bf16* Kb = Kt + (size_t)bh * NN * HD;
    const __bf16* Vb = Vt + (size_t)bh * HD * NN;

    // Q fragments (16x64 -> two 16x32 A-frags), loaded once.
    v16bf qa0 = load_afrag(Qb, HD, row0, 0, lane);
    v16bf qa1 = load_afrag(Qb, HD, row0, 32, lane);

    v8f o[4] = {v8f{}, v8f{}, v8f{}, v8f{}};
    float mr[8], lr[8];
#pragma unroll
    for (int i = 0; i < 8; ++i) { mr[i] = -1e30f; lr[i] = 0.0f; }

    int col = lane & 15;
    int ro  = (lane >> 4) * 8;             // C-layout row offset for this half-wave

    // prime K fragments for j = 0
    v16bf kf0 = load_bfrag(Kb, HD, 0, 0, lane);
    v16bf kf1 = load_bfrag(Kb, HD, 0, 32, lane);
    v16bf kf2 = load_bfrag(Kb, HD, 16, 0, lane);
    v16bf kf3 = load_bfrag(Kb, HD, 16, 32, lane);

#pragma unroll 2
    for (int j = 0; j < NN; j += 32) {
        // V fragments for this step: issue early, consumed at the bottom.
        v16bf vf[4];
#pragma unroll
        for (int t4 = 0; t4 < 4; ++t4) vf[t4] = load_bfrag(Vb, NN, 16 * t4, j, lane);

        // ---- scores for 32 keys: two 16x16 C tiles ----
        v8f s0 = {}, s1 = {};
        s0 = wmma_bf16(qa0, kf0, s0);
        s0 = wmma_bf16(qa1, kf1, s0);
        s1 = wmma_bf16(qa0, kf2, s1);
        s1 = wmma_bf16(qa1, kf3, s1);

        // ---- prefetch next step's K fragments (overlap softmax below) ----
        int jn = (j + 32 < NN) ? (j + 32) : j;   // last iter: harmless reload
        kf0 = load_bfrag(Kb, HD, jn, 0, lane);
        kf1 = load_bfrag(Kb, HD, jn, 32, lane);
        kf2 = load_bfrag(Kb, HD, jn + 16, 0, lane);
        kf3 = load_bfrag(Kb, HD, jn + 16, 32, lane);

        // ---- online softmax (row stats live per half-wave lane) ----
        float cf[8];
#pragma unroll
        for (int i = 0; i < 8; ++i) {
            float a0 = s0[i] * SCALE;
            float a1 = s1[i] * SCALE;
            float t = fmaxf(a0, a1);
            t = fmaxf(t, __shfl_xor(t, 1, 32));
            t = fmaxf(t, __shfl_xor(t, 2, 32));
            t = fmaxf(t, __shfl_xor(t, 4, 32));
            t = fmaxf(t, __shfl_xor(t, 8, 32));
            float mn = fmaxf(mr[i], t);
            float c  = __expf(mr[i] - mn);
            float p0 = __expf(a0 - mn);
            float p1 = __expf(a1 - mn);
            float rs = p0 + p1;
            rs += __shfl_xor(rs, 1, 32);
            rs += __shfl_xor(rs, 2, 32);
            rs += __shfl_xor(rs, 4, 32);
            rs += __shfl_xor(rs, 8, 32);
            lr[i] = lr[i] * c + rs;
            mr[i] = mn;
            cf[i] = c;
            s0[i] = p0;
            s1[i] = p1;
        }

        // ---- P: C-layout -> LDS [16][32] bf16 -> A-layout frag ----
#pragma unroll
        for (int i = 0; i < 8; ++i) {
            Pl[(i + ro) * 32 + col]      = f2bf(s0[i]);
            Pl[(i + ro) * 32 + 16 + col] = f2bf(s1[i]);
        }
        __syncthreads();   // single-wave WG: DS in-order; barrier is ~free

        // ---- rescale running O, then PV ----
#pragma unroll
        for (int t4 = 0; t4 < 4; ++t4)
#pragma unroll
            for (int i = 0; i < 8; ++i) o[t4][i] *= cf[i];

        v16bf pa = load_afrag(Pl, 32, 0, 0, lane);
#pragma unroll
        for (int t4 = 0; t4 < 4; ++t4)
            o[t4] = wmma_bf16(pa, vf[t4], o[t4]);
        __syncthreads();
    }

    // ---- finalize and write context (bf16, [B,N,D] so out-proj reads rows) ----
    float inv[8];
#pragma unroll
    for (int i = 0; i < 8; ++i) inv[i] = 1.0f / lr[i];
#pragma unroll
    for (int t4 = 0; t4 < 4; ++t4)
#pragma unroll
        for (int i = 0; i < 8; ++i) {
            size_t r = (size_t)b * NN + row0 + i + ro;
            ctx[r * DD + h * HD + 16 * t4 + col] = f2bf(o[t4][i] * inv[i]);
        }
}

// ---------------------------------------------------------------------------
// Kernel 4: out projection.  out[4096,1024] f32 = ctx_bf16 @ w_out + b_out.
// ---------------------------------------------------------------------------
__global__ __launch_bounds__(256) void out_gemm_kernel(const __bf16* __restrict__ Ctx,
                                                       const __bf16* __restrict__ Wt,
                                                       const float* __restrict__ bias,
                                                       float* __restrict__ out) {
    int wid  = blockIdx.x * 8 + (threadIdx.x >> 5);
    int lane = threadIdx.x & 31;
    int mt = wid >> 4;              // 256 row-tiles
    int nt = wid & 15;              // 16 col-tiles of 64
    int row0 = mt * 16;
    int n0   = nt * 64;

    v8f acc[4] = {v8f{}, v8f{}, v8f{}, v8f{}};
    gemm_16x64_pipelined(Ctx, Wt, row0, n0, lane, acc);

    int col   = lane & 15;
    int hflag = lane >> 4;
#pragma unroll
    for (int t = 0; t < 4; ++t) {
        int   c  = n0 + 16 * t + col;
        float bv = bias[c];
#pragma unroll
        for (int i = 0; i < 8; ++i) {
            size_t r = (size_t)row0 + i + 8 * hflag;
            out[r * DD + c] = acc[t][i] + bv;
        }
    }
}

// ---------------------------------------------------------------------------
// Launcher.  Workspace layout (48 MB total):
//   [0,8M)   x bf16        [8M,14M)  w_qkv^T bf16   [14M,16M) w_out^T bf16
//   [16M,24M) Q [B,H,N,HD] [24M,32M) K [B,H,N,HD]   [32M,40M) V^T [B,H,HD,N]
//   [40M,48M) ctx [B,N,D] bf16
// ---------------------------------------------------------------------------
extern "C" void kernel_launch(void* const* d_in, const int* in_sizes, int n_in,
                              void* d_out, int out_size, void* d_ws, size_t ws_size,
                              hipStream_t stream) {
    const float* x     = (const float*)d_in[0];
    const float* wqkv  = (const float*)d_in[1];
    const float* bqkv  = (const float*)d_in[2];
    const float* wout  = (const float*)d_in[3];
    const float* bout  = (const float*)d_in[4];
    float*       out   = (float*)d_out;

    char* ws = (char*)d_ws;
    __bf16* xbf   = (__bf16*)(ws);
    __bf16* wqkvT = (__bf16*)(ws + ((size_t)8  << 20));
    __bf16* woutT = (__bf16*)(ws + ((size_t)14 << 20));
    __bf16* Q     = (__bf16*)(ws + ((size_t)16 << 20));
    __bf16* K     = (__bf16*)(ws + ((size_t)24 << 20));
    __bf16* Vt    = (__bf16*)(ws + ((size_t)32 << 20));
    __bf16* ctx   = (__bf16*)(ws + ((size_t)40 << 20));

    // 1) convert / transpose to bf16
    {
        size_t total = NX + NW1 + NW2;                      // 8388608
        int blocks = (int)((total + 255) / 256);            // 32768
        convert_kernel<<<blocks, 256, 0, stream>>>(x, wqkv, wout, xbf, wqkvT, woutT);
    }
    // 2) QKV projection: 256 * 48 waves, 8 waves/block
    qkv_gemm_kernel<<<1536, 256, 0, stream>>>(xbf, wqkvT, bqkv, Q, K, Vt);
    // 3) flash attention: one wave per (b,h,q-tile) = 4096 waves
    attn_kernel<<<4096, 32, 0, stream>>>(Q, K, Vt, ctx);
    // 4) output projection: 256 * 16 waves, 8 waves/block
    out_gemm_kernel<<<512, 256, 0, stream>>>(ctx, woutT, bout, out);
}
